// MultiHeadAttention_16767552323903
// MI455X (gfx1250) — matrix operand order repaired and checked by reading, not compiled, not run on hardware
//
#include <hip/hip_runtime.h>

typedef __attribute__((ext_vector_type(16))) __bf16 bf16x16;
typedef __attribute__((ext_vector_type(8)))  __bf16 bf16x8;
typedef __attribute__((ext_vector_type(8)))  float  f32x8;

#define S_LEN 2048
#define DMODEL 1024
#define NH 16
#define HD 64

// Finite-input RNE f32 -> bf16 (inputs here are always finite: uniform weights,
// normal x, exp() outputs, finite accumulations).
__device__ __forceinline__ unsigned short f32_to_bf16_rne(float f) {
  unsigned int u = __float_as_uint(f);
  u += 0x7fffu + ((u >> 16) & 1u);
  return (unsigned short)(u >> 16);
}

__device__ __forceinline__ f32x8 zero8() {
  f32x8 z;
#pragma unroll
  for (int i = 0; i < 8; ++i) z[i] = 0.0f;
  return z;
}

union AFragU { bf16x16 v; bf16x8 h[2]; };

// A-fragment (16-bit, 16x32 MxK): lane L holds row M=L&15,
// K = half*16 + 0..15 contiguous (matches B fragments).
// Two b128 loads straight into the fragment register halves.
__device__ __forceinline__ bf16x16 load_a_frag(const unsigned short* rowPtr, int half) {
  AFragU u;
  u.h[0] = *reinterpret_cast<const bf16x8*>(rowPtr + half * 16);
  u.h[1] = *reinterpret_cast<const bf16x8*>(rowPtr + half * 16 + 8);
  return u.v;
}

// B-fragment (16-bit, 32x16 KxN): lane L holds col N=L&15, K = half*16 + 0..15 contiguous.
__device__ __forceinline__ bf16x16 load_b_frag(const unsigned short* p) {
  return *reinterpret_cast<const bf16x16*>(p);
}

__global__ void cvt_f32_bf16(const float* __restrict__ src,
                             unsigned short* __restrict__ dst, int n) {
  int i = blockIdx.x * blockDim.x + threadIdx.x;
  if (i < n) dst[i] = f32_to_bf16_rne(src[i]);
}

// C = A[M,1024] @ W[1024,1024]^T   (W row-major [out,in] == B^T row-major -> contiguous B frags)
// mode 0: write bf16 as [B*H, S, HD]  (Q / K)
// mode 1: write bf16 as [B*H, HD, S]  (V transposed)
__global__ __launch_bounds__(256)
void gemm_qkv_wmma(const unsigned short* __restrict__ A,
                   const unsigned short* __restrict__ W,
                   unsigned short* __restrict__ Out, int mode) {
  const int lane = threadIdx.x & 31;
  const int wave = threadIdx.x >> 5;
  const int half = lane >> 4;
  const int l16  = lane & 15;
  const int m0 = blockIdx.x * 128 + (wave & 3) * 32;
  const int n0 = blockIdx.y * 128 + (wave >> 2) * 64;

  const unsigned short* arow[2];
  const unsigned short* wrow[4];
#pragma unroll
  for (int mi = 0; mi < 2; ++mi) arow[mi] = A + (size_t)(m0 + mi * 16 + l16) * DMODEL;
#pragma unroll
  for (int ni = 0; ni < 4; ++ni) wrow[ni] = W + (size_t)(n0 + ni * 16 + l16) * DMODEL + half * 16;

  f32x8 acc[2][4];
#pragma unroll
  for (int i = 0; i < 2; ++i)
#pragma unroll
    for (int j = 0; j < 4; ++j) acc[i][j] = zero8();

#pragma unroll 2
  for (int kk = 0; kk < DMODEL; kk += 32) {
    // prefetch next k-tile into cache while WMMAs drain (global_prefetch_b8)
    __builtin_prefetch(arow[0] + kk + 32, 0, 3);
    __builtin_prefetch(wrow[0] + kk + 32, 0, 3);
    bf16x16 afrag[2];
#pragma unroll
    for (int mi = 0; mi < 2; ++mi) afrag[mi] = load_a_frag(arow[mi] + kk, half);
    bf16x16 bfrag[4];
#pragma unroll
    for (int ni = 0; ni < 4; ++ni) bfrag[ni] = load_b_frag(wrow[ni] + kk);
#pragma unroll
    for (int mi = 0; mi < 2; ++mi)
#pragma unroll
      for (int ni = 0; ni < 4; ++ni)
        acc[mi][ni] = __builtin_amdgcn_wmma_f32_16x16x32_bf16(
            false, afrag[mi], false, bfrag[ni], (short)0, acc[mi][ni], false, false);
  }

#pragma unroll
  for (int mi = 0; mi < 2; ++mi)
#pragma unroll
    for (int ni = 0; ni < 4; ++ni)
#pragma unroll
      for (int r = 0; r < 8; ++r) {
        const int gm = m0 + mi * 16 + r + 8 * half;   // 0..8191 = b*2048 + s
        const int gn = n0 + ni * 16 + l16;            // 0..1023 = h*64 + d
        const int b = gm >> 11, s = gm & 2047;
        const int h = gn >> 6,  d = gn & 63;
        size_t idx;
        if (mode == 0) idx = (((size_t)(b * NH + h) * S_LEN) + s) * HD + d;
        else           idx = (((size_t)(b * NH + h) * HD) + d) * S_LEN + s;
        Out[idx] = f32_to_bf16_rne(acc[mi][ni][r]);
      }
}

// Flash attention: grid (S/128, B*H), 8 waves, each wave owns 16 q-rows.
__global__ __launch_bounds__(256)
void flash_attn_wmma(const unsigned short* __restrict__ Q,   // [B*H, S, 64]
                     const unsigned short* __restrict__ K,   // [B*H, S, 64]
                     const unsigned short* __restrict__ VT,  // [B*H, 64, S]
                     unsigned short* __restrict__ CTX) {     // [B*S, 1024] bf16
  // Per-wave P tile; row stride 48 ushorts = 96B keeps 8-element chunks 16B-aligned.
  __shared__ __align__(16) unsigned short plds[8][16][48];

  const int lane = threadIdx.x & 31;
  const int wave = threadIdx.x >> 5;
  const int half = lane >> 4;
  const int l16  = lane & 15;
  const int bh   = blockIdx.y;
  const int q0   = blockIdx.x * 128 + wave * 16;
  const size_t qkBase = (size_t)bh * S_LEN * HD;
  const size_t vBase  = (size_t)bh * HD * S_LEN;

  bf16x16 qa[2];
#pragma unroll
  for (int c = 0; c < 2; ++c)
    qa[c] = load_a_frag(Q + qkBase + (size_t)(q0 + l16) * HD + c * 32, half);

  f32x8 oacc[4];
#pragma unroll
  for (int t = 0; t < 4; ++t) oacc[t] = zero8();
  float mrow[8], lrow[8];
#pragma unroll
  for (int r = 0; r < 8; ++r) { mrow[r] = -__builtin_inff(); lrow[r] = 0.0f; }

  const int q_hi = q0 + 15;
  const int kend = blockIdx.x * 128 + 128;
  const float scale = 0.125f;   // 1/sqrt(64)

  for (int k0 = 0; k0 < kend; k0 += 32) {
    const bool active = (k0 <= q_hi);   // wave-uniform -> EXEC all-1s inside
    float alpha[8];
    if (active) {
      f32x8 sacc[2];
#pragma unroll
      for (int t = 0; t < 2; ++t) {
        sacc[t] = zero8();
#pragma unroll
        for (int c = 0; c < 2; ++c) {
          bf16x16 kb = load_b_frag(
              K + qkBase + (size_t)(k0 + t * 16 + l16) * HD + c * 32 + half * 16);
          sacc[t] = __builtin_amdgcn_wmma_f32_16x16x32_bf16(
              false, qa[c], false, kb, (short)0, sacc[t], false, false);
        }
      }
      // scale + causal mask
#pragma unroll
      for (int t = 0; t < 2; ++t)
#pragma unroll
        for (int r = 0; r < 8; ++r) {
          const int qq = q0 + r + 8 * half;
          const int kc = k0 + t * 16 + l16;
          float s = sacc[t][r] * scale;
          sacc[t][r] = (kc <= qq) ? s : -__builtin_inff();
        }
      // online softmax (rows live across 16-lane groups; xor<16 stays in group)
#pragma unroll
      for (int r = 0; r < 8; ++r) {
        float lm = fmaxf(sacc[0][r], sacc[1][r]);
#pragma unroll
        for (int off = 1; off < 16; off <<= 1) lm = fmaxf(lm, __shfl_xor(lm, off));
        const float mnew = fmaxf(mrow[r], lm);
        const float p0 = __expf(sacc[0][r] - mnew);
        const float p1 = __expf(sacc[1][r] - mnew);
        float rs = p0 + p1;
#pragma unroll
        for (int off = 1; off < 16; off <<= 1) rs += __shfl_xor(rs, off);
        alpha[r] = __expf(mrow[r] - mnew);
        lrow[r] = lrow[r] * alpha[r] + rs;
        mrow[r] = mnew;
        sacc[0][r] = p0; sacc[1][r] = p1;   // reuse as P storage
      }
      // C-layout -> A-layout transpose through per-wave LDS slice
#pragma unroll
      for (int t = 0; t < 2; ++t)
#pragma unroll
        for (int r = 0; r < 8; ++r)
          plds[wave][r + 8 * half][t * 16 + l16] = f32_to_bf16_rne(sacc[t][r]);
      // rescale running output
#pragma unroll
      for (int t = 0; t < 4; ++t)
#pragma unroll
        for (int r = 0; r < 8; ++r) oacc[t][r] *= alpha[r];
    }
    __syncthreads();
    if (active) {
      AFragU pu;
      pu.h[0] = *reinterpret_cast<const bf16x8*>(&plds[wave][l16][half * 16]);
      pu.h[1] = *reinterpret_cast<const bf16x8*>(&plds[wave][l16][half * 16 + 8]);
#pragma unroll
      for (int t = 0; t < 4; ++t) {
        bf16x16 vb = load_b_frag(
            VT + vBase + (size_t)(t * 16 + l16) * S_LEN + k0 + half * 16);
        oacc[t] = __builtin_amdgcn_wmma_f32_16x16x32_bf16(
            false, pu.v, false, vb, (short)0, oacc[t], false, false);
      }
    }
    __syncthreads();
  }

  const int b = bh >> 4, h = bh & 15;
#pragma unroll
  for (int r = 0; r < 8; ++r) {
    const float invl = 1.0f / lrow[r];
    const int qq = q0 + r + 8 * half;
    const size_t row = (size_t)(b * S_LEN + qq);
#pragma unroll
    for (int t = 0; t < 4; ++t) {
      const int col = h * HD + t * 16 + l16;
      CTX[row * DMODEL + col] = f32_to_bf16_rne(oacc[t][r] * invl);
    }
  }
}

// Out = CTX[8192,1024] @ Wo^T + bo   (f32 output)
__global__ __launch_bounds__(256)
void gemm_out_proj(const unsigned short* __restrict__ A,
                   const unsigned short* __restrict__ W,
                   const float* __restrict__ bias,
                   float* __restrict__ Out) {
  const int lane = threadIdx.x & 31;
  const int wave = threadIdx.x >> 5;
  const int half = lane >> 4;
  const int l16  = lane & 15;
  const int m0 = blockIdx.x * 128 + (wave & 3) * 32;
  const int n0 = blockIdx.y * 128 + (wave >> 2) * 64;

  const unsigned short* arow[2];
  const unsigned short* wrow[4];
#pragma unroll
  for (int mi = 0; mi < 2; ++mi) arow[mi] = A + (size_t)(m0 + mi * 16 + l16) * DMODEL;
#pragma unroll
  for (int ni = 0; ni < 4; ++ni) wrow[ni] = W + (size_t)(n0 + ni * 16 + l16) * DMODEL + half * 16;

  f32x8 acc[2][4];
#pragma unroll
  for (int i = 0; i < 2; ++i)
#pragma unroll
    for (int j = 0; j < 4; ++j) acc[i][j] = zero8();

#pragma unroll 2
  for (int kk = 0; kk < DMODEL; kk += 32) {
    __builtin_prefetch(arow[0] + kk + 32, 0, 3);
    __builtin_prefetch(wrow[0] + kk + 32, 0, 3);
    bf16x16 afrag[2];
#pragma unroll
    for (int mi = 0; mi < 2; ++mi) afrag[mi] = load_a_frag(arow[mi] + kk, half);
    bf16x16 bfrag[4];
#pragma unroll
    for (int ni = 0; ni < 4; ++ni) bfrag[ni] = load_b_frag(wrow[ni] + kk);
#pragma unroll
    for (int mi = 0; mi < 2; ++mi)
#pragma unroll
      for (int ni = 0; ni < 4; ++ni)
        acc[mi][ni] = __builtin_amdgcn_wmma_f32_16x16x32_bf16(
            false, afrag[mi], false, bfrag[ni], (short)0, acc[mi][ni], false, false);
  }

#pragma unroll
  for (int mi = 0; mi < 2; ++mi)
#pragma unroll
    for (int ni = 0; ni < 4; ++ni)
#pragma unroll
      for (int r = 0; r < 8; ++r) {
        const int gm = m0 + mi * 16 + r + 8 * half;
        const int gn = n0 + ni * 16 + l16;
        Out[(size_t)gm * DMODEL + gn] = acc[mi][ni][r] + bias[gn];
      }
}

extern "C" void kernel_launch(void* const* d_in, const int* in_sizes, int n_in,
                              void* d_out, int out_size, void* d_ws, size_t ws_size,
                              hipStream_t stream) {
  const float* x  = (const float*)d_in[0];
  const float* Wq = (const float*)d_in[1];
  const float* Wk = (const float*)d_in[2];
  const float* Wv = (const float*)d_in[3];
  const float* Wo = (const float*)d_in[4];
  const float* bo = (const float*)d_in[5];
  float* out = (float*)d_out;

  char* ws = (char*)d_ws;
  const size_t MB = 1024u * 1024u;
  unsigned short* Xb  = (unsigned short*)(ws);            // 16 MB, later reused as CTX
  unsigned short* WQb = (unsigned short*)(ws + 16 * MB);  // 2 MB each
  unsigned short* WKb = (unsigned short*)(ws + 18 * MB);
  unsigned short* WVb = (unsigned short*)(ws + 20 * MB);
  unsigned short* WOb = (unsigned short*)(ws + 22 * MB);
  unsigned short* Qb  = (unsigned short*)(ws + 24 * MB);  // 16 MB each
  unsigned short* Kb  = (unsigned short*)(ws + 40 * MB);
  unsigned short* VTb = (unsigned short*)(ws + 56 * MB);
  unsigned short* CTXb = Xb;  // x no longer needed once QKV GEMMs finished

  const int XN = 4 * S_LEN * DMODEL;   // 8,388,608
  const int WN = DMODEL * DMODEL;      // 1,048,576
  cvt_f32_bf16<<<(XN + 255) / 256, 256, 0, stream>>>(x, Xb, XN);
  cvt_f32_bf16<<<(WN + 255) / 256, 256, 0, stream>>>(Wq, WQb, WN);
  cvt_f32_bf16<<<(WN + 255) / 256, 256, 0, stream>>>(Wk, WKb, WN);
  cvt_f32_bf16<<<(WN + 255) / 256, 256, 0, stream>>>(Wv, WVb, WN);
  cvt_f32_bf16<<<(WN + 255) / 256, 256, 0, stream>>>(Wo, WOb, WN);

  dim3 gg(64, 8);   // M=8192/128, N=1024/128
  gemm_qkv_wmma<<<gg, 256, 0, stream>>>(Xb, WQb, Qb, 0);
  gemm_qkv_wmma<<<gg, 256, 0, stream>>>(Xb, WKb, Kb, 0);
  gemm_qkv_wmma<<<gg, 256, 0, stream>>>(Xb, WVb, VTb, 1);

  flash_attn_wmma<<<dim3(S_LEN / 128, 4 * NH), 256, 0, stream>>>(Qb, Kb, VTb, CTXb);

  gemm_out_proj<<<gg, 256, 0, stream>>>(CTXb, WOb, bo, out);
}
